// InferenceOnceImpl_63591285784654
// MI455X (gfx1250) — matrix operand order repaired and checked by reading, not compiled, not run on hardware
//
#include <hip/hip_runtime.h>
#include <math.h>

// ---------------------------------------------------------------------------
// MI455X (gfx1250) implementation of the streaming Demucs denoiser step.
// Bandwidth-bound workload (~76MB weights/call @ 23.3 TB/s -> ~3.3us floor):
// all matmul-shaped work (encoder stride-4 convs via im2col, 1x1 GLU convs,
// LSTM GEMVs) runs on v_wmma_f32_16x16x32_f16 (wave32, one 16x16 tile/wave),
// with b128 vector loads and no divergent guards in the inner loop (OOB rows/
// cols are clamped, not zeroed -- their products land in never-stored lanes).
// ---------------------------------------------------------------------------

typedef __attribute__((ext_vector_type(16))) _Float16 v16h;
typedef __attribute__((ext_vector_type(8)))  float    v8f;

#define PI_F 3.14159265358979323846f

__device__ __forceinline__ float sigmoidf_(float x) { return 1.f / (1.f + expf(-x)); }

// ---------------- resampling kernel table (hanning-windowed sinc) ----------
__global__ void k_resample_kernel(float* rk) {
  int i = threadIdx.x;
  if (i < 112) {
    // win = hanning(225)[1::2] ; t = (-55.5 + i) * pi
    float win = 0.5f - 0.5f * cosf(2.f * PI_F * (float)(2 * i + 1) / 224.f);
    float t = (-55.5f + (float)i) * PI_F;
    float s = (t == 0.f) ? 1.f : (sinf(t) / t);
    rk[i] = s * win;
  }
}

// ---------------- variance of concat(process_buf[256:], x) -----------------
__global__ void k_variance(const float* __restrict__ pbuf,
                           const float* __restrict__ x, float* sv) {
  __shared__ float red[256];
  int t = threadIdx.x;
  float s = 0.f;
  for (int i = t; i < 661; i += 256) {
    float v = (i < 405) ? pbuf[256 + i] : x[i - 405];
    s += v * v;
  }
  red[t] = s;
  __syncthreads();
  for (int o = 128; o > 0; o >>= 1) {
    if (t < o) red[t] += red[t + o];
    __syncthreads();
  }
  if (t == 0) sv[0] = sqrtf(red[0] / 661.f);
}

// ---------------- frame917 = concat(resample_in, buf/(floor+sqrtvar)) ------
__global__ void k_frame(const float* __restrict__ pbuf, const float* __restrict__ x,
                        const float* __restrict__ rin, const float* __restrict__ sv,
                        float* __restrict__ f917) {
  int i = blockIdx.x * blockDim.x + threadIdx.x;
  if (i >= 917) return;
  if (i < 256) { f917[i] = rin[i]; return; }
  int j = i - 256;
  float v = (j < 405) ? pbuf[256 + j] : x[j - 405];
  f917[i] = v / (0.001f + sv[0]);
}

// ---------------- upsample2: out[2t]=x[t], out[2t+1]=sinc interp -----------
__global__ void k_upsample(const float* __restrict__ x, const float* __restrict__ rk,
                           float* __restrict__ out, int T) {
  int i = blockIdx.x * blockDim.x + threadIdx.x;
  if (i >= 2 * T) return;
  int t = i >> 1;
  if ((i & 1) == 0) { out[i] = x[t]; return; }
  float acc = 0.f;
  int base = t + 1 - 56;
  #pragma unroll 4
  for (int j = 0; j < 112; j++) {
    int idx = base + j;
    if (idx >= 0 && idx < T) acc += x[idx] * rk[j];
  }
  out[i] = acc;
}

// ---------------- downsample2 ----------------------------------------------
__global__ void k_downsample(const float* __restrict__ x, const float* __restrict__ rk,
                             float* __restrict__ out, int Le) {
  int T2 = Le >> 1;
  int i = blockIdx.x * blockDim.x + threadIdx.x;
  if (i >= T2) return;
  float acc = 0.f;
  int base = i - 56;
  #pragma unroll 4
  for (int j = 0; j < 112; j++) {
    int m = base + j;
    if (m >= 0 && m < T2) acc += x[2 * m + 1] * rk[j];
  }
  out[i] = 0.5f * (x[2 * i] + acc);
}

// ---------------- generic WMMA GEMM: C = act(A(MxK) * B(KxN) + bias [+C]) --
// bmode==0 : B dense (K x N) row-major
// bmode==1 : im2col of stride-4 k=8 conv: K=cin*8, N=Lout,
//            B[k][n] = X[(k>>3)*Lin + 4n + (k&7)]
// One wave computes a 16x16 output tile with v_wmma_f32_16x16x32_f16.
// Layouts here (contiguous k, A and B alike): A lane holds row (lane&15), elems 0..7 = K[k0+16hi..+7],
// elems 8..15 = K[k0+16hi+8..+15] (two contiguous float4x2 runs -> b128 loads);
// B lane holds col (lane&15), elems e = K[k0+16hi+e]; D rows m0+8hi+r.
__global__ void k_wmma_gemm(const float* __restrict__ A, const float* __restrict__ B,
                            const float* __restrict__ bias, float* __restrict__ C,
                            int M, int N, int K, int bmode, int Lin,
                            int act, int accum) {
  int wave = (blockIdx.x * blockDim.x + threadIdx.x) >> 5;
  int lane = threadIdx.x & 31;
  int Mt = (M + 15) >> 4;
  int Nt = (N + 15) >> 4;
  if (wave >= Mt * Nt) return;
  int mt = wave % Mt, nt = wave / Mt;
  int m0 = mt << 4, n0 = nt << 4;
  int hi = lane >> 4;                 // half-wave select
  int mrow = m0 + (lane & 15);        // A row owned by this lane
  int ncol = n0 + (lane & 15);        // B/C column owned by this lane
  // Clamp instead of zero: products from clamped (duplicate) rows/cols only
  // reach D rows/cols that the epilogue never writes back.
  int mrowc = (mrow < M) ? mrow : (M - 1);
  int ncolc = (ncol < N) ? ncol : (N - 1);
  const float* Arow = A + (size_t)mrowc * K;

  v8f acc = {};
  for (int k0 = 0; k0 < K; k0 += 32) {
    v16h a, b;
    if (k0 + 32 <= K) {               // wave-uniform fast path: no guards
      // --- A: two contiguous 8-float runs -> 4x global_load_b128 ---
      const float4* pa0 = (const float4*)(Arow + k0 + 16 * hi);
      const float4* pa1 = (const float4*)(Arow + k0 + 16 * hi + 8);
      float4 a0 = pa0[0], a1 = pa0[1];
      float4 a2 = pa1[0], a3 = pa1[1];
      a[0]=(_Float16)a0.x; a[1]=(_Float16)a0.y; a[2]=(_Float16)a0.z; a[3]=(_Float16)a0.w;
      a[4]=(_Float16)a1.x; a[5]=(_Float16)a1.y; a[6]=(_Float16)a1.z; a[7]=(_Float16)a1.w;
      a[8]=(_Float16)a2.x; a[9]=(_Float16)a2.y; a[10]=(_Float16)a2.z; a[11]=(_Float16)a2.w;
      a[12]=(_Float16)a3.x; a[13]=(_Float16)a3.y; a[14]=(_Float16)a3.z; a[15]=(_Float16)a3.w;
      int kb = k0 + 16 * hi;
      if (bmode == 0) {
        #pragma unroll
        for (int e = 0; e < 16; ++e)
          b[e] = (_Float16)B[(size_t)(kb + e) * N + ncolc];
      } else {
        // im2col: elems 0..7 and 8..15 are contiguous 8-float runs too
        const float* xp0 = B + (size_t)(kb >> 3) * Lin + (ncolc << 2);
        const float* xp1 = xp0 + Lin;
        float4 b0 = ((const float4*)xp0)[0], b1 = ((const float4*)xp0)[1];
        float4 b2 = ((const float4*)xp1)[0], b3 = ((const float4*)xp1)[1];
        b[0]=(_Float16)b0.x; b[1]=(_Float16)b0.y; b[2]=(_Float16)b0.z; b[3]=(_Float16)b0.w;
        b[4]=(_Float16)b1.x; b[5]=(_Float16)b1.y; b[6]=(_Float16)b1.z; b[7]=(_Float16)b1.w;
        b[8]=(_Float16)b2.x; b[9]=(_Float16)b2.y; b[10]=(_Float16)b2.z; b[11]=(_Float16)b2.w;
        b[12]=(_Float16)b3.x; b[13]=(_Float16)b3.y; b[14]=(_Float16)b3.z; b[15]=(_Float16)b3.w;
      }
    } else {
      // K tail (K%32 != 0): clamped loads + select, no exec divergence
      #pragma unroll
      for (int v = 0; v < 8; ++v) {
        int base = 2 * v + 16 * hi;
        int k = k0 + base;
        int kc0 = (k < K) ? k : (K - 1);
        int kc1 = (k + 1 < K) ? (k + 1) : (K - 1);
        float x0 = Arow[kc0], x1 = Arow[kc1];
        a[2 * v]     = (_Float16)((k < K) ? x0 : 0.f);
        a[2 * v + 1] = (_Float16)((k + 1 < K) ? x1 : 0.f);
      }
      int kb = k0 + 16 * hi;
      #pragma unroll
      for (int e = 0; e < 16; ++e) {
        int k = kb + e;
        int kc = (k < K) ? k : (K - 1);
        float xv = (bmode == 0)
                 ? B[(size_t)kc * N + ncolc]
                 : B[(size_t)(kc >> 3) * Lin + (ncolc << 2) + (kc & 7)];
        b[e] = (_Float16)((k < K) ? xv : 0.f);
      }
    }
    acc = __builtin_amdgcn_wmma_f32_16x16x32_f16(
        /*neg_a=*/false, a, /*neg_b=*/false, b,
        /*c_mod=*/(short)0, acc, /*reuse_a=*/false, /*reuse_b=*/false);
  }
  // --- epilogue: lanes0-15 rows m0..m0+7, lanes16-31 rows m0+8..m0+15 ---
  int mbase = m0 + hi * 8;
  if (ncol < N) {
    #pragma unroll
    for (int r = 0; r < 8; ++r) {
      int m = mbase + r;
      if (m < M) {
        float v = acc[r];
        if (bias) v += bias[m];
        if (accum) v += C[(size_t)m * N + ncol];
        if (act == 1) v = fmaxf(v, 0.f);
        C[(size_t)m * N + ncol] = v;
      }
    }
  }
}

// ---------------- GLU: out(C,L) = t[0:C] * sigmoid(t[C:2C]) ----------------
__global__ void k_glu(const float* __restrict__ t, float* __restrict__ o, int C, int L) {
  int i = blockIdx.x * blockDim.x + threadIdx.x;
  if (i >= C * L) return;
  o[i] = t[i] * sigmoidf_(t[C * L + i]);
}

// ---------------- LSTM cell nonlinearity -----------------------------------
__global__ void k_lstm(const float* __restrict__ g, const float* __restrict__ cprev,
                       float* __restrict__ out) {
  int t = blockIdx.x * blockDim.x + threadIdx.x;
  if (t >= 768) return;
  float ig = sigmoidf_(g[t]);
  float fg = sigmoidf_(g[768 + t]);
  float gg = tanhf(g[1536 + t]);
  float og = sigmoidf_(g[2304 + t]);
  float cn = fg * cprev[t] + ig * gg;
  out[t] = og * tanhf(cn);
}

// ---------------- dst(C,Ld) += src(C,Ls)[:, off:off+Ld] --------------------
__global__ void k_add2d(float* __restrict__ dst, const float* __restrict__ src,
                        int C, int Ld, int Ls, int off) {
  int i = blockIdx.x * blockDim.x + threadIdx.x;
  if (i >= C * Ld) return;
  int c = i / Ld, n = i % Ld;
  dst[i] += src[c * Ls + off + n];
}

// ---------------- conv_transpose1d, stride 4, k=8 --------------------------
// Y(Cout, 4L+4) = convT(X(Cin,L), W(Cin,Cout,8)) + bias. Exactly 2 taps/out.
__global__ void k_convt(const float* __restrict__ X, const float* __restrict__ W,
                        const float* __restrict__ bias, float* __restrict__ Y,
                        int Cin, int Cout, int L) {
  int Tout = 4 * L + 4;
  int i = blockIdx.x * blockDim.x + threadIdx.x;
  if (i >= Cout * Tout) return;
  int co = i / Tout, t = i % Tout;
  int j1 = t & 3;
  int i1 = t >> 2;      // tap with kernel index j1
  int i2 = i1 - 1;      // tap with kernel index j1+4
  bool v1 = (i1 < L);
  bool v2 = (i2 >= 0);
  float acc = bias[co];
  for (int ci = 0; ci < Cin; ci++) {
    const float* wb = W + (ci * Cout + co) * 8;
    const float* xb = X + ci * L;
    if (v1) acc += wb[j1]     * xb[i1];
    if (v2) acc += wb[j1 + 4] * xb[i2];
  }
  Y[i] = acc;
}

// ---------------- extra tail: init=copy last4, else extra[:, :4]+=tail-bt --
__global__ void k_tail(float* __restrict__ extra, const float* __restrict__ Y,
                       const float* __restrict__ bt, int C, int Tout, int Lext, int init) {
  int i = blockIdx.x * blockDim.x + threadIdx.x;
  if (i >= C * 4) return;
  int c = i >> 2, j = i & 3;
  float v = Y[c * Tout + (Tout - 4) + j];
  if (init) extra[c * 4 + j] = v;
  else      extra[c * Lext + j] += v - bt[c];
}

// ---------------- trim last 4 samples per row ------------------------------
__global__ void k_trim(const float* __restrict__ Y, float* __restrict__ X, int C, int Tout) {
  int Lx = Tout - 4;
  int i = blockIdx.x * blockDim.x + threadIdx.x;
  if (i >= C * Lx) return;
  int c = i / Lx, n = i % Lx;
  X[i] = Y[c * Tout + n];
}

__global__ void k_relu(float* p, int n) {
  int i = blockIdx.x * blockDim.x + threadIdx.x;
  if (i < n) p[i] = fmaxf(p[i], 0.f);
}

__global__ void k_copy(float* __restrict__ d, const float* __restrict__ s, int n) {
  int i = blockIdx.x * blockDim.x + threadIdx.x;
  if (i < n) d[i] = s[i];
}

__global__ void k_final(const float* __restrict__ d2, const float* __restrict__ sv,
                        float* __restrict__ out) {
  int i = blockIdx.x * blockDim.x + threadIdx.x;
  if (i < 256) out[i] = d2[64 + i] * sv[0];
}

// ---------------------------------------------------------------------------
// workspace layout (float offsets)
enum {
  OFF_RK    = 0,        // 112
  OFF_SV    = 112,      // 4
  OFF_F917  = 128,      // 917
  OFF_UP1   = 1056,     // 1834
  OFF_F3668 = 2944,     // 3668
  OFF_CONVB = 6656,     // 28608  (stride4 conv out, max 48*596)
  OFF_TMP   = 35328,    // 57216  (pre-GLU, max 96*596)
  OFF_SKIP0 = 92544,    // 48*596
  OFF_SKIP1 = 121152,   // 96*148
  OFF_SKIP2 = 135360,   // 192*36
  OFF_SKIP3 = 142272,   // 384*8
  OFF_SKIP4 = 145344,   // 768
  OFF_GATES = 146112,   // 3072
  OFF_LOUT0 = 149184,   // 768
  OFF_LOUT1 = 149952,   // 768   (decoder x at entry)
  OFF_DXA   = 150720,   // 16384
  OFF_DXB   = 167104,   // 16384
  OFF_GX    = 183488,   // 16384
  OFF_EXA   = 199872,   // 20480
  OFF_EXB   = 220352,   // 20480
  OFF_GE    = 240832,   // 20480
  OFF_YBUF  = 261312,   // 16384
  OFF_PAD   = 277696,   // 2644
  OFF_DOWN1 = 280352,   // 1322
  OFF_DOWN2 = 281680,   // 661
};

static inline int nb(int n) { return (n + 255) / 256; }

extern "C" void kernel_launch(void* const* d_in, const int* in_sizes, int n_in,
                              void* d_out, int out_size, void* d_ws, size_t ws_size,
                              hipStream_t stream) {
  (void)in_sizes; (void)n_in; (void)out_size; (void)ws_size;

  const float* x      = (const float*)d_in[0];
  const float* pbuf   = (const float*)d_in[1];
  const float* rin    = (const float*)d_in[2];
  const float* rout   = (const float*)d_in[3];
  const float* hstate = (const float*)d_in[4];   // (2,1,768)
  const float* cstate = (const float*)d_in[5];   // (2,1,768)

  const float* encW[5][4];
  for (int l = 0; l < 5; l++)
    for (int k = 0; k < 4; k++) encW[l][k] = (const float*)d_in[6 + 4 * l + k];
  const float* decW[5][4];
  for (int l = 0; l < 5; l++)
    for (int k = 0; k < 4; k++) decW[l][k] = (const float*)d_in[26 + 4 * l + k];
  const float* lstmW[2][4];
  for (int l = 0; l < 2; l++)
    for (int k = 0; k < 4; k++) lstmW[l][k] = (const float*)d_in[46 + 4 * l + k];

  float* w = (float*)d_ws;
  float* rk = w + OFF_RK;
  float* sv = w + OFF_SV;

  auto gemm = [&](const float* A, const float* B, const float* bias, float* C,
                  int M, int N, int K, int bmode, int Lin, int act, int accum) {
    int tiles = ((M + 15) / 16) * ((N + 15) / 16);
    int blocks = (tiles + 3) / 4;                     // 4 waves (128 thr) / block
    k_wmma_gemm<<<blocks, 128, 0, stream>>>(A, B, bias, C, M, N, K, bmode, Lin, act, accum);
  };

  // ---- front-end: variance, normalize, 4x upsample ----
  k_resample_kernel<<<1, 128, 0, stream>>>(rk);
  k_variance<<<1, 256, 0, stream>>>(pbuf, x, sv);
  k_frame<<<nb(917), 256, 0, stream>>>(pbuf, x, rin, sv, w + OFF_F917);
  k_upsample<<<nb(1834), 256, 0, stream>>>(w + OFF_F917, rk, w + OFF_UP1, 917);
  k_upsample<<<nb(3668), 256, 0, stream>>>(w + OFF_UP1, rk, w + OFF_F3668, 1834);
  const float* f2388 = w + OFF_F3668 + 1024;   // f3668[:, 1024:1024+2388]

  // ---- encoder: 5x (stride-4 conv + relu, 1x1 conv + GLU), all WMMA ----
  float* skips[5] = { w + OFF_SKIP0, w + OFF_SKIP1, w + OFF_SKIP2, w + OFF_SKIP3, w + OFF_SKIP4 };
  int cin[5]  = { 1, 48, 96, 192, 384 };
  int cout[5] = { 48, 96, 192, 384, 768 };
  int Ls[5];
  {
    const float* prev = f2388;
    int Lin = 2388;
    for (int l = 0; l < 5; l++) {
      int Lout = (Lin - 8) / 4 + 1;
      Ls[l] = Lout;
      gemm(encW[l][0], prev, encW[l][1], w + OFF_CONVB,
           cout[l], Lout, cin[l] * 8, /*bmode=*/1, Lin, /*relu=*/1, 0);
      gemm(encW[l][2], w + OFF_CONVB, encW[l][3], w + OFF_TMP,
           2 * cout[l], Lout, cout[l], 0, 0, 0, 0);
      k_glu<<<nb(cout[l] * Lout), 256, 0, stream>>>(w + OFF_TMP, skips[l], cout[l], Lout);
      prev = skips[l];
      Lin = Lout;
    }
  }

  // ---- 2-layer LSTM on the 768-vector (WMMA GEMVs) ----
  {
    const float* lin = skips[4];
    float* louts[2] = { w + OFF_LOUT0, w + OFF_LOUT1 };
    for (int l = 0; l < 2; l++) {
      gemm(lstmW[l][0], lin,            lstmW[l][2], w + OFF_GATES, 3072, 1, 768, 0, 0, 0, 0);
      gemm(lstmW[l][1], hstate + l*768, lstmW[l][3], w + OFF_GATES, 3072, 1, 768, 0, 0, 0, 1);
      k_lstm<<<3, 256, 0, stream>>>(w + OFF_GATES, cstate + l * 768, louts[l]);
      lin = louts[l];
    }
  }

  // ---- decoder with "extra" tail path ----
  int dch[5] = { 768, 384, 192, 96, 48 };
  int dco[5] = { 384, 192, 96, 48, 1 };
  float* dx = w + OFF_LOUT1;          // x starts as lstm output (768,1)
  int Lx = 1;
  float* xbufs[2] = { w + OFF_DXA, w + OFF_DXB };
  float* ebufs[2] = { w + OFF_EXA, w + OFF_EXB };
  int xb = 0, eb = 0;
  float* exCur = nullptr;
  int Lext = 0;

  for (int idx = 0; idx < 5; idx++) {
    const float* sk = skips[4 - idx];
    int Lsk = Ls[4 - idx];
    int C = dch[idx], Co = dco[idx];

    // x += skip[:, :Lx]; x = GLU(conv1x1(x))
    k_add2d<<<nb(C * Lx), 256, 0, stream>>>(dx, sk, C, Lx, Lsk, 0);
    gemm(decW[idx][0], dx, decW[idx][1], w + OFF_TMP, 2 * C, Lx, C, 0, 0, 0, 0);
    k_glu<<<nb(C * Lx), 256, 0, stream>>>(w + OFF_TMP, w + OFF_GX, C, Lx);

    // extra path (exists from idx>=1)
    if (exCur) {
      k_add2d<<<nb(C * Lext), 256, 0, stream>>>(exCur, sk, C, Lext, Lsk, Lx);
      gemm(decW[idx][0], exCur, decW[idx][1], w + OFF_TMP, 2 * C, Lext, C, 0, 0, 0, 0);
      k_glu<<<nb(C * Lext), 256, 0, stream>>>(w + OFF_TMP, w + OFF_GE, C, Lext);
      float* exNext = ebufs[eb ^ 1];
      int Te = 4 * Lext + 4;
      k_convt<<<nb(Co * Te), 256, 0, stream>>>(w + OFF_GE, decW[idx][2], decW[idx][3],
                                               exNext, C, Co, Lext);
      exCur = exNext; eb ^= 1; Lext = Te;
    }

    // x = convT(x)
    int Ty = 4 * Lx + 4;
    k_convt<<<nb(Co * Ty), 256, 0, stream>>>(w + OFF_GX, decW[idx][2], decW[idx][3],
                                             w + OFF_YBUF, C, Co, Lx);
    if (!exCur) {
      exCur = ebufs[0]; Lext = 4;
      k_tail<<<nb(Co * 4), 256, 0, stream>>>(exCur, w + OFF_YBUF, decW[idx][3], Co, Ty, 4, 1);
    } else {
      k_tail<<<nb(Co * 4), 256, 0, stream>>>(exCur, w + OFF_YBUF, decW[idx][3], Co, Ty, Lext, 0);
    }
    float* xn = xbufs[xb]; xb ^= 1;
    k_trim<<<nb(Co * (Ty - 4)), 256, 0, stream>>>(w + OFF_YBUF, xn, Co, Ty);
    dx = xn; Lx = Ty - 4;

    if (idx != 4) {
      k_relu<<<nb(Co * Lx), 256, 0, stream>>>(dx, Co * Lx);
      k_relu<<<nb(Co * Lext), 256, 0, stream>>>(exCur, Co * Lext);
    }
  }

  // ---- back-end: concat, 2x downsample, rescale ----
  float* pad = w + OFF_PAD;
  k_copy<<<nb(256), 256, 0, stream>>>(pad, rout, 256);
  k_copy<<<nb(1024), 256, 0, stream>>>(pad + 256, dx, 1024);
  k_copy<<<nb(1364), 256, 0, stream>>>(pad + 1280, exCur, 1364);
  k_downsample<<<nb(1322), 256, 0, stream>>>(pad, rk, w + OFF_DOWN1, 2644);
  k_downsample<<<nb(661), 256, 0, stream>>>(w + OFF_DOWN1, rk, w + OFF_DOWN2, 1322);
  k_final<<<1, 256, 0, stream>>>(w + OFF_DOWN2, sv, (float*)d_out);
}
